// SelfAttn_29678224015696
// MI455X (gfx1250) — matrix operand order repaired and checked by reading, not compiled, not run on hardware
//
#include <hip/hip_runtime.h>
#include <hip/hip_bf16.h>

// Problem constants (match the reference)
#define DDIM 1024
#define NHEAD 16
#define DHEAD 64
#define TSEQ 2048
#define BBATCH 2
#define BT (BBATCH * TSEQ)

typedef __bf16 v16bf __attribute__((ext_vector_type(16)));
typedef float  v8f   __attribute__((ext_vector_type(8)));

union FragAB { v16bf v; unsigned int u[8]; };
union FragC  { v8f   v; float        f[8]; };

__device__ __forceinline__ unsigned short f2bf(float x) {
  unsigned int u = __float_as_uint(x);
  u += 0x7fffu + ((u >> 16) & 1u);          // round-to-nearest-even
  return (unsigned short)(u >> 16);
}
__device__ __forceinline__ float bf2f(unsigned short h) {
  return __uint_as_float(((unsigned int)h) << 16);
}
__device__ __forceinline__ unsigned int pack_bf16(float lo, float hi) {
  return (unsigned int)f2bf(lo) | ((unsigned int)f2bf(hi) << 16);
}

// ---------------------------------------------------------------- convert
__global__ void cvt_f32_bf16(const float* __restrict__ in,
                             unsigned short* __restrict__ out, int n) {
  int i = blockIdx.x * blockDim.x + threadIdx.x;
  int stride = gridDim.x * blockDim.x;
  for (; i < n; i += stride) out[i] = f2bf(in[i]);
}

// ---------------------------------------------------------------- GEMM
// C[M,N] = A[M,K] * B[K,N], bf16 in, f32 accumulate.
// Block tile 256x128, 8 waves (4x2), wave tile 64x64 -> 16 WMMA per k-step.
template <bool OUT_F32>
__global__ __launch_bounds__(256) void gemm_bf16_256x128(
    const unsigned short* __restrict__ A, const unsigned short* __restrict__ B,
    void* __restrict__ Cout, int M, int N, int K) {
  __shared__ unsigned short As[256 * 34];  // [m][k], stride 34 halves (pad)
  __shared__ unsigned short Bs[128 * 34];  // [n][k] transposed, stride 34

  const int tid = threadIdx.x;
  const int lane = tid & 31, wid = tid >> 5;
  const int lm = lane & 15, lh = lane >> 4;
  const int wm = wid >> 1, wn = wid & 1;   // 4x2 wave grid
  const int m0 = blockIdx.y * 256, n0 = blockIdx.x * 128;

  FragC acc[4][4];
#pragma unroll
  for (int mi = 0; mi < 4; mi++)
#pragma unroll
    for (int ni = 0; ni < 4; ni++)
#pragma unroll
      for (int v = 0; v < 8; v++) acc[mi][ni].f[v] = 0.f;

  for (int k0 = 0; k0 < K; k0 += 32) {
    {  // A tile: 256 rows x 32 halves; one 64B row per thread
      const uint4* src = (const uint4*)(A + (size_t)(m0 + tid) * K + k0);
      uint4 q0 = src[0], q1 = src[1], q2 = src[2], q3 = src[3];
      unsigned int* dst = (unsigned int*)(As + tid * 34);
      dst[0] = q0.x;  dst[1] = q0.y;  dst[2]  = q0.z;  dst[3]  = q0.w;
      dst[4] = q1.x;  dst[5] = q1.y;  dst[6]  = q1.z;  dst[7]  = q1.w;
      dst[8] = q2.x;  dst[9] = q2.y;  dst[10] = q2.z;  dst[11] = q2.w;
      dst[12] = q3.x; dst[13] = q3.y; dst[14] = q3.z;  dst[15] = q3.w;
    }
    {  // B tile: 32 k-rows x 128 n-cols, scatter transposed into Bs[n][k]
      int k = tid >> 3, n = (tid & 7) * 16;
      const uint4* src = (const uint4*)(B + (size_t)(k0 + k) * N + n0 + n);
      uint4 q0 = src[0], q1 = src[1];
      unsigned short h[16];
      ((uint4*)h)[0] = q0; ((uint4*)h)[1] = q1;
#pragma unroll
      for (int i = 0; i < 16; i++) Bs[(n + i) * 34 + k] = h[i];
    }
    __syncthreads();

    FragAB a[4], b[4];
#pragma unroll
    for (int mi = 0; mi < 4; mi++) {
      int r = wm * 64 + mi * 16 + lm;
#pragma unroll
      for (int v = 0; v < 8; v++) {
        // A k order here (contiguous, matches B): lanes 0-15 K=0..15, lanes 16-31 K=16..31
        int kk = lh * 16 + 2 * v;
        a[mi].u[v] = *(const unsigned int*)(As + r * 34 + kk);
      }
    }
#pragma unroll
    for (int ni = 0; ni < 4; ni++) {
      int c = wn * 64 + ni * 16 + lm;
#pragma unroll
      for (int v = 0; v < 8; v++) {
        // B k order here (contiguous, matches A): lane half -> K 0-15 / 16-31, vgpr -> K pair
        int kk = lh * 16 + 2 * v;
        b[ni].u[v] = *(const unsigned int*)(Bs + c * 34 + kk);
      }
    }
#pragma unroll
    for (int mi = 0; mi < 4; mi++)
#pragma unroll
      for (int ni = 0; ni < 4; ni++)
        acc[mi][ni].v = __builtin_amdgcn_wmma_f32_16x16x32_bf16(
            false, a[mi].v, false, b[ni].v, (short)0, acc[mi][ni].v, false, false);
    __syncthreads();
  }

#pragma unroll
  for (int mi = 0; mi < 4; mi++)
#pragma unroll
    for (int ni = 0; ni < 4; ni++)
#pragma unroll
      for (int v = 0; v < 8; v++) {
        int row = m0 + wm * 64 + mi * 16 + lh * 8 + v;  // C layout: row=half*8+v
        int col = n0 + wn * 64 + ni * 16 + lm;
        if (OUT_F32)
          ((float*)Cout)[(size_t)row * N + col] = acc[mi][ni].f[v];
        else
          ((unsigned short*)Cout)[(size_t)row * N + col] = f2bf(acc[mi][ni].f[v]);
      }
}

// ---------------------------------------------------------------- RoPE + split
// qkv: [BT, 3*D] bf16 -> Q/K/V: [B, NH, T, DH] bf16, RoPE on Q,K (f32 math).
__global__ void rope_split_kernel(const unsigned short* __restrict__ qkv,
                                  unsigned short* __restrict__ Q,
                                  unsigned short* __restrict__ K,
                                  unsigned short* __restrict__ V) {
  int idx = blockIdx.x * blockDim.x + threadIdx.x;  // BT*NH*32 threads
  int d = idx & 31;
  int row = idx >> 5;
  int h = row & (NHEAD - 1);
  int bt = row >> 4;
  int t = bt & (TSEQ - 1);
  int b = bt >> 11;  // /TSEQ
  size_t src = (size_t)bt * (3 * DDIM) + h * DHEAD;
  float q1 = bf2f(qkv[src + d]);
  float q2 = bf2f(qkv[src + d + 32]);
  float k1 = bf2f(qkv[src + DDIM + d]);
  float k2 = bf2f(qkv[src + DDIM + d + 32]);
  float theta = powf(10000.0f, -(float)d * (1.0f / 32.0f));
  float ang = (float)t * theta;
  float sn, cs;
  sincosf(ang, &sn, &cs);
  size_t dst = (((size_t)b * NHEAD + h) * TSEQ + t) * DHEAD;
  Q[dst + d]      = f2bf(q1 * cs - q2 * sn);
  Q[dst + d + 32] = f2bf(q1 * sn + q2 * cs);
  K[dst + d]      = f2bf(k1 * cs - k2 * sn);
  K[dst + d + 32] = f2bf(k1 * sn + k2 * cs);
  V[dst + d]      = qkv[src + 2 * DDIM + d];
  V[dst + d + 32] = qkv[src + 2 * DDIM + d + 32];
}

// ---------------------------------------------------------------- attention
// One block = 128 queries of one (b,h). 8 waves x 16 queries.
// Transposed formulation: S^T = K * Q^T (query on the lane index) so the
// online softmax is in-lane (one shfl_xor(16) per max / per sum), and
// O^T = V^T * P^T where the P^T B-fragment is assembled in registers
// (4 packed-dword half-exchanges, no LDS staging for P).
__global__ __launch_bounds__(256) void attn_kernel(
    const unsigned short* __restrict__ Q, const unsigned short* __restrict__ Kh,
    const unsigned short* __restrict__ V, unsigned short* __restrict__ Y) {
  __shared__ unsigned short Ks[32 * 66];  // [key][dim], stride 66
  __shared__ unsigned short Vs[64 * 34];  // [dim][key] (transposed)

  const int tid = threadIdx.x, lane = tid & 31, wid = tid >> 5;
  const int lm = lane & 15, lh = lane >> 4;
  const int qt = blockIdx.x & 15, bh = blockIdx.x >> 4;
  const int b = bh >> 4, h = bh & 15;
  const int q0 = qt * 128;
  const int qbase = q0 + wid * 16;
  const int q = qbase + lm;  // this lane's query (column of S^T / O^T)
  const size_t head_off = (size_t)bh * TSEQ * DHEAD;

  // Q^T B-fragments: lane -> query, vgpr -> head-dim pairs; kept in VGPRs
  FragAB qb[2];
#pragma unroll
  for (int s = 0; s < 2; s++)
#pragma unroll
    for (int v = 0; v < 8; v++) {
      int d = s * 32 + lh * 16 + 2 * v;
      qb[s].u[v] =
          *(const unsigned int*)(Q + head_off + (size_t)q * DHEAD + d);
    }

  float m_run = -__builtin_inff(), l_run = 0.f;
  FragC O[4];  // O^T: rows = dims (f*16 + half*8 + v), cols = queries
#pragma unroll
  for (int f = 0; f < 4; f++)
#pragma unroll
    for (int v = 0; v < 8; v++) O[f].f[v] = 0.f;

  for (int j0 = 0; j0 < q0 + 128; j0 += 32) {
    __syncthreads();
    {  // cooperative K/V tile load: 32 keys x 64 dims each
      int key = tid >> 3, d0 = (tid & 7) * 8;
      uint4 kq = *(const uint4*)(Kh + head_off + (size_t)(j0 + key) * DHEAD + d0);
      unsigned int* kdst = (unsigned int*)(Ks + key * 66 + d0);
      kdst[0] = kq.x; kdst[1] = kq.y; kdst[2] = kq.z; kdst[3] = kq.w;
      uint4 vq = *(const uint4*)(V + head_off + (size_t)(j0 + key) * DHEAD + d0);
      unsigned short hh[8];
      *(uint4*)hh = vq;
#pragma unroll
      for (int i = 0; i < 8; i++) Vs[(d0 + i) * 34 + key] = hh[i];  // transpose
    }
    __syncthreads();

    // S^T = K * Q^T : 32 keys x 16 queries -> 2 C-frags (rows = keys)
    FragC S[2];
#pragma unroll
    for (int g = 0; g < 2; g++) {
#pragma unroll
      for (int v = 0; v < 8; v++) S[g].f[v] = 0.f;
#pragma unroll
      for (int s = 0; s < 2; s++) {
        FragAB ka;  // A layout: lane -> key row, vgpr/half -> dim
#pragma unroll
        for (int v = 0; v < 8; v++) {
          int kk = s * 32 + lh * 16 + 2 * v;
          ka.u[v] = *(const unsigned int*)(Ks + (g * 16 + lm) * 66 + kk);
        }
        S[g].v = __builtin_amdgcn_wmma_f32_16x16x32_bf16(
            false, ka.v, false, qb[s].v, (short)0, S[g].v, false, false);
      }
    }

    // in-lane online softmax for this lane's query
    float p[2][8];
    float mx = -__builtin_inff();
#pragma unroll
    for (int g = 0; g < 2; g++)
#pragma unroll
      for (int v = 0; v < 8; v++) {
        int key = j0 + g * 16 + lh * 8 + v;
        float xx = S[g].f[v] * 0.125f;  // 1/sqrt(64)
        if (key > q) xx = -__builtin_inff();
        p[g][v] = xx;
        mx = fmaxf(mx, xx);
      }
    mx = fmaxf(mx, __shfl_xor(mx, 16, 32));  // combine key halves
    float mnew = fmaxf(m_run, mx);
    float sc = __expf(m_run - mnew);
    float ssum = 0.f;
#pragma unroll
    for (int g = 0; g < 2; g++)
#pragma unroll
      for (int v = 0; v < 8; v++) {
        float e = __expf(p[g][v] - mnew);
        p[g][v] = e;
        ssum += e;
      }
    ssum += __shfl_xor(ssum, 16, 32);
    l_run = l_run * sc + ssum;
    m_run = mnew;

    // assemble P^T B-fragment: lane -> query, vgpr -> key pair.
    // lane half owns half the key rows; swap the other half with lane^16.
    unsigned int d0[4], d1[4], recv[4];
#pragma unroll
    for (int v = 0; v < 4; v++) {
      d0[v] = pack_bf16(p[0][2 * v], p[0][2 * v + 1]);  // keys  lh*8 + 2v
      d1[v] = pack_bf16(p[1][2 * v], p[1][2 * v + 1]);  // keys 16+lh*8+2v
    }
#pragma unroll
    for (int v = 0; v < 4; v++) {
      unsigned int send = lh ? d0[v] : d1[v];
      recv[v] = __shfl_xor(send, 16, 32);
    }
    FragAB pb;
#pragma unroll
    for (int v = 0; v < 4; v++) {
      pb.u[v]     = lh ? recv[v] : d0[v];  // k = half*16 + 2v
      pb.u[4 + v] = lh ? d1[v]   : recv[v];
    }

    // O^T = O^T*sc + V^T * P^T   (A = V^T: lane -> dim row, vgpr -> key)
#pragma unroll
    for (int f = 0; f < 4; f++) {
      FragAB va;
#pragma unroll
      for (int v = 0; v < 8; v++) {
        int kk = lh * 16 + 2 * v;
        va.u[v] = *(const unsigned int*)(Vs + (f * 16 + lm) * 34 + kk);
      }
      FragC oc;
#pragma unroll
      for (int v = 0; v < 8; v++) oc.f[v] = O[f].f[v] * sc;
      O[f].v = __builtin_amdgcn_wmma_f32_16x16x32_bf16(
          false, va.v, false, pb.v, (short)0, oc.v, false, false);
    }
  }

  // normalize + packed 16B stores: lane owns query q, rows = dims
  float inv = 1.0f / l_run;
#pragma unroll
  for (int f = 0; f < 4; f++) {
    unsigned int w[4];
#pragma unroll
    for (int j = 0; j < 4; j++)
      w[j] = pack_bf16(O[f].f[2 * j] * inv, O[f].f[2 * j + 1] * inv);
    uint4 out; out.x = w[0]; out.y = w[1]; out.z = w[2]; out.w = w[3];
    *(uint4*)(Y + (size_t)(b * TSEQ + q) * DDIM + h * DHEAD + f * 16 + lh * 8) = out;
  }
}

// ---------------------------------------------------------------- launch
extern "C" void kernel_launch(void* const* d_in, const int* in_sizes, int n_in,
                              void* d_out, int out_size, void* d_ws,
                              size_t ws_size, hipStream_t stream) {
  (void)in_sizes; (void)n_in; (void)out_size; (void)ws_size;
  const float* x = (const float*)d_in[0];
  const float* Wqkv = (const float*)d_in[1];
  const float* Wo = (const float*)d_in[2];

  unsigned short* ws = (unsigned short*)d_ws;
  size_t off = 0;
  unsigned short* Xb = ws + off;    off += (size_t)BT * DDIM;
  unsigned short* Wqkvb = ws + off; off += (size_t)DDIM * 3 * DDIM;
  unsigned short* Wob = ws + off;   off += (size_t)DDIM * DDIM;
  unsigned short* qkvb = ws + off;  off += (size_t)BT * 3 * DDIM;
  unsigned short* Qb = ws + off;    off += (size_t)BBATCH * NHEAD * TSEQ * DHEAD;
  unsigned short* Kb = ws + off;    off += (size_t)BBATCH * NHEAD * TSEQ * DHEAD;
  unsigned short* Vb = ws + off;    off += (size_t)BBATCH * NHEAD * TSEQ * DHEAD;
  unsigned short* Yb = ws + off;    off += (size_t)BT * DDIM;

  cvt_f32_bf16<<<2048, 256, 0, stream>>>(x, Xb, BT * DDIM);
  cvt_f32_bf16<<<2048, 256, 0, stream>>>(Wqkv, Wqkvb, DDIM * 3 * DDIM);
  cvt_f32_bf16<<<1024, 256, 0, stream>>>(Wo, Wob, DDIM * DDIM);

  gemm_bf16_256x128<false><<<dim3(3 * DDIM / 128, BT / 256), 256, 0, stream>>>(
      Xb, Wqkvb, (void*)qkvb, BT, 3 * DDIM, DDIM);

  rope_split_kernel<<<(BT * NHEAD * 32) / 256, 256, 0, stream>>>(qkvb, Qb, Kb, Vb);

  attn_kernel<<<BBATCH * NHEAD * (TSEQ / 128), 256, 0, stream>>>(Qb, Kb, Vb, Yb);

  gemm_bf16_256x128<true><<<dim3(DDIM / 128, BT / 256), 256, 0, stream>>>(
      Yb, Wob, d_out, BT, DDIM, DDIM);
}
